// Attention_63917703299638
// MI455X (gfx1250) — matrix operand order repaired and checked by reading, not compiled, not run on hardware
//
#include <hip/hip_runtime.h>
#include <hip/hip_bf16.h>
#include <math.h>
#include <stdint.h>

typedef __attribute__((ext_vector_type(16))) __bf16 v16bf;
typedef __attribute__((ext_vector_type(8)))  __bf16 v8bf;
typedef __attribute__((ext_vector_type(8)))  float  v8f;

#define NQ     4096
#define DMODEL 512
#define HHEADS 8
#define DHD    64
#define HD     512        // H*DH
#define MMEM   4
#define JTOT   4100       // N + M
#define JPAD   4160
#define WINSZ  512

// ---------------------------------------------------------------------------
// fp32 -> bf16 convert
__global__ void k_cvt(const float* __restrict__ s, __bf16* __restrict__ d, int n) {
    int i = blockIdx.x * blockDim.x + threadIdx.x;
    if (i < n) d[i] = (__bf16)s[i];
}

// transpose + convert: W[K][Nc] row-major -> Wt[Nc][K] bf16
__global__ void k_cvtT(const float* __restrict__ W, __bf16* __restrict__ Wt, int K, int Nc) {
    int i = blockIdx.x * blockDim.x + threadIdx.x;
    if (i >= K * Nc) return;
    int n = i / K, k = i % K;
    Wt[i] = (__bf16)W[(size_t)k * Nc + n];
}

// ---------------------------------------------------------------------------
// C[M,Nc] (f32) = A[M,K] (bf16, row-major) @ Bt[Nc,K] (bf16, row-major = B^T)
// block: 256 threads = 8 waves; tile 128 rows x 64 cols; wave = 16 rows.
// B tile (64 cols x 32 k = 4KB) staged once per block via async-to-LDS DMA,
// double-buffered and overlapped with WMMA compute (ASYNCcnt pipeline).
__global__ __launch_bounds__(256) void k_gemm_bf16(const __bf16* __restrict__ A,
                                                   const __bf16* __restrict__ Bt,
                                                   float* __restrict__ C,
                                                   int Mrows, int Ncols, int K) {
    __shared__ __bf16 bs[2][64 * 32];
    const int tid  = threadIdx.x;
    const int wave = tid >> 5;
    const int lane = tid & 31;
    const int lo16 = lane & 15;
    const int half = lane >> 4;
    const int rowBase = blockIdx.x * 128 + wave * 16;
    const int colBase = blockIdx.y * 64;

    // cooperative staging map: 256 threads x 16B cover 64 rows x 64B
    const int srow   = tid >> 2;   // B column (0..63)
    const int schunk = tid & 3;    // 16B chunk within the 64B row
    const __bf16* bsrc = Bt + (size_t)(colBase + srow) * K + schunk * 8;
    const uint32_t lds0 = (uint32_t)(uintptr_t)(&bs[0][0]) + srow * 64 + schunk * 16;
    const uint32_t lds1 = (uint32_t)(uintptr_t)(&bs[1][0]) + srow * 64 + schunk * 16;

    v8f acc[4] = {};
    const __bf16* arow = A + (size_t)(rowBase + lo16) * K;
    const int S = K >> 5;          // 32-wide k-steps

    // prologue: stage k-step 0 into buffer 0
    {
        uint64_t ga = (uint64_t)(uintptr_t)(bsrc);
        asm volatile("global_load_async_to_lds_b128 %0, %1, off"
                     :: "v"(lds0), "v"(ga) : "memory");
    }
    for (int i = 0; i < S; ++i) {
        const int k0 = i << 5;
        if (i + 1 < S) {
            uint64_t ga = (uint64_t)(uintptr_t)(bsrc + (k0 + 32));
            uint32_t ld = ((i + 1) & 1) ? lds1 : lds0;
            asm volatile("global_load_async_to_lds_b128 %0, %1, off"
                         :: "v"(ld), "v"(ga) : "memory");
            asm volatile("s_wait_asynccnt 0x1" ::: "memory");
        } else {
            asm volatile("s_wait_asynccnt 0x0" ::: "memory");
        }
        __syncthreads();                      // stage i visible to all waves

        if (i + 2 < S) __builtin_prefetch(arow + k0 + 64, 0, 0);

        // A fragment (global): lane<16 -> K {k0..+15}; lane>=16 -> K {k0+16..+31}
        const __bf16* ap = arow + k0 + half * 16;
        v8bf a_lo = *(const v8bf*)(ap);
        v8bf a_hi = *(const v8bf*)(ap + 8);
        v16bf afrag;
#pragma unroll
        for (int e = 0; e < 8; ++e) { afrag[e] = a_lo[e]; afrag[e + 8] = a_hi[e]; }

        const __bf16* bb = bs[i & 1];
#pragma unroll
        for (int t = 0; t < 4; ++t) {
            v16bf bfrag = *(const v16bf*)(bb + (t * 16 + lo16) * 32 + half * 16);
            acc[t] = __builtin_amdgcn_wmma_f32_16x16x32_bf16(false, afrag, false, bfrag,
                                                             (short)0, acc[t], false, false);
        }
        __syncthreads();                      // all waves done before overwrite
    }
#pragma unroll
    for (int t = 0; t < 4; ++t)
#pragma unroll
        for (int r = 0; r < 8; ++r) {
            int row = rowBase + r + half * 8;
            int col = colBase + t * 16 + lo16;
            C[(size_t)row * Ncols + col] = acc[t][r];
        }
}

// ---------------------------------------------------------------------------
// l2norm + rotary on Q/K, repack: Qh[h][n][d], Kf[h][4+n][d], Vt[h][d][4+n]
__global__ void k_qk_post(const float* __restrict__ Qf, const float* __restrict__ Kr,
                          const float* __restrict__ Vf, const float* __restrict__ freqs,
                          __bf16* __restrict__ Qh, __bf16* __restrict__ Kf,
                          __bf16* __restrict__ Vt) {
    int idx = blockIdx.x * blockDim.x + threadIdx.x;
    if (idx >= NQ * HHEADS) return;
    int n = idx / HHEADS, h = idx % HHEADS;
    const float* qp = Qf + (size_t)n * HD + h * DHD;
    const float* kp = Kr + (size_t)n * HD + h * DHD;
    const float* vp = Vf + (size_t)n * HD + h * DHD;
    const float* fr = freqs + (size_t)n * DHD;

    float qs = 0.f, ks = 0.f;
    for (int d = 0; d < DHD; ++d) { float q = qp[d], k = kp[d]; qs += q * q; ks += k * k; }
    float qi = rsqrtf(fmaxf(qs, 1e-24f));
    float ki = rsqrtf(fmaxf(ks, 1e-24f));

    __bf16* qo = Qh + ((size_t)h * NQ + n) * DHD;
    __bf16* ko = Kf + ((size_t)h * JPAD + MMEM + n) * DHD;
    for (int d = 0; d < DHD; ++d) {
        float c = __cosf(fr[d]), s = __sinf(fr[d]);
        float q = qp[d] * qi, k = kp[d] * ki;
        float qr = (d < 32) ? -qp[d + 32] * qi : qp[d - 32] * qi;
        float kr = (d < 32) ? -kp[d + 32] * ki : kp[d - 32] * ki;
        qo[d] = (__bf16)(q * c + qr * s);
        ko[d] = (__bf16)(k * c + kr * s);
    }
    __bf16* vo = Vt + (size_t)h * DHD * JPAD;
    for (int d = 0; d < DHD; ++d) vo[(size_t)d * JPAD + MMEM + n] = (__bf16)vp[d];
}

// memory tokens -> Kf rows 0..3 (l2-normed), Vt cols 0..3
__global__ void k_mem(const float* __restrict__ mem_k, const float* __restrict__ mem_v,
                      __bf16* __restrict__ Kf, __bf16* __restrict__ Vt) {
    int idx = blockIdx.x * blockDim.x + threadIdx.x;
    if (idx >= HHEADS * MMEM) return;
    int h = idx >> 2, m = idx & 3;
    const float* kp = mem_k + ((size_t)h * MMEM + m) * DHD;
    const float* vp = mem_v + ((size_t)h * MMEM + m) * DHD;
    float s = 0.f;
    for (int d = 0; d < DHD; ++d) s += kp[d] * kp[d];
    float inv = rsqrtf(fmaxf(s, 1e-24f));
    __bf16* ko = Kf + ((size_t)h * JPAD + m) * DHD;
    for (int d = 0; d < DHD; ++d) ko[d] = (__bf16)(kp[d] * inv);
    __bf16* vo = Vt + (size_t)h * DHD * JPAD;
    for (int d = 0; d < DHD; ++d) vo[(size_t)d * JPAD + m] = (__bf16)vp[d];
}

// ---------------------------------------------------------------------------
// windowed flash attention + sigmoid gating. grid (N/64, H), 128 thr = 4 waves.
__global__ __launch_bounds__(128) void k_attn(const __bf16* __restrict__ Qh,
                                              const __bf16* __restrict__ Kf,
                                              const __bf16* __restrict__ Vt,
                                              const float* __restrict__ Gf,
                                              const float* __restrict__ bg,
                                              __bf16* __restrict__ Ag) {
    __shared__ __bf16 lds_p[4][16 * 32];
    const int h = blockIdx.y;
    const int blk = blockIdx.x;
    const int wave = threadIdx.x >> 5;
    const int lane = threadIdx.x & 31;
    const int lo16 = lane & 15;
    const int half = lane >> 4;
    const int qbase = blk * 64 + wave * 16;

    // Q A-fragments (d 0..31 and 32..63)
    const __bf16* Qrow = Qh + ((size_t)h * NQ + qbase + lo16) * DHD;
    v16bf aq[2];
#pragma unroll
    for (int t = 0; t < 2; ++t) {
        const __bf16* qp = Qrow + t * 32 + half * 16;
        v8bf lo = *(const v8bf*)qp;
        v8bf hi = *(const v8bf*)(qp + 8);
#pragma unroll
        for (int e = 0; e < 8; ++e) { aq[t][e] = lo[e]; aq[t][e + 8] = hi[e]; }
    }

    float mrun[8], lrun[8];
    v8f acc[4] = {};
#pragma unroll
    for (int r = 0; r < 8; ++r) { mrun[r] = -3.0e38f; lrun[r] = 0.f; }

    int lo = blk * 64 + MMEM - WINSZ; if (lo < 0) lo = 0; lo &= ~31;
    int hiEnd = blk * 64 + 63 + MMEM + WINSZ + 1; if (hiEnd > JTOT) hiEnd = JTOT;

    const __bf16* Kb = Kf + (size_t)h * JPAD * DHD;
    const __bf16* Vb = Vt + (size_t)h * DHD * JPAD;
    __bf16* pl = lds_p[wave];

    for (int jb = lo; jb < hiEnd; jb += 32) {
        // ---- S = (Q K^T) * 10 with window mask ----
        v8f S[2];
#pragma unroll
        for (int tile = 0; tile < 2; ++tile) {
            int jk = jb + tile * 16 + lo16;
            const __bf16* krow = Kb + (size_t)jk * DHD;
            v8f c = {};
#pragma unroll
            for (int t = 0; t < 2; ++t) {
                v16bf bk = *(const v16bf*)(krow + t * 32 + half * 16);
                c = __builtin_amdgcn_wmma_f32_16x16x32_bf16(false, aq[t], false, bk,
                                                            (short)0, c, false, false);
            }
#pragma unroll
            for (int r = 0; r < 8; ++r) {
                int p = qbase + r + half * 8 + MMEM;   // global query position
                int dist = p - jk;
                bool ok = (jk < JTOT) && (dist <= WINSZ) && (dist >= -WINSZ);
                c[r] = ok ? c[r] * 10.0f : -3.0e38f;
            }
            S[tile] = c;
        }
        // ---- online softmax (rows live across 16-lane halves) ----
        float alpha[8];
#pragma unroll
        for (int r = 0; r < 8; ++r) {
            float mx = fmaxf(S[0][r], S[1][r]);
#pragma unroll
            for (int m = 8; m >= 1; m >>= 1) mx = fmaxf(mx, __shfl_xor(mx, m, 32));
            float mnew = fmaxf(mrun[r], mx);
            alpha[r] = __expf(mrun[r] - mnew);
            mrun[r] = mnew;
            float p0 = __expf(S[0][r] - mnew);
            float p1 = __expf(S[1][r] - mnew);
            S[0][r] = p0; S[1][r] = p1;
            float rs = p0 + p1;
#pragma unroll
            for (int m = 8; m >= 1; m >>= 1) rs += __shfl_xor(rs, m, 32);
            lrun[r] = lrun[r] * alpha[r] + rs;
        }
#pragma unroll
        for (int t = 0; t < 4; ++t)
#pragma unroll
            for (int r = 0; r < 8; ++r) acc[t][r] *= alpha[r];

        // ---- P (C-layout) -> LDS -> A-layout fragment ----
#pragma unroll
        for (int tile = 0; tile < 2; ++tile)
#pragma unroll
            for (int r = 0; r < 8; ++r)
                pl[(r + half * 8) * 32 + tile * 16 + lo16] = (__bf16)S[tile][r];
        __syncthreads();
        v16bf pf;
        {
            const __bf16* pp = pl + lo16 * 32 + half * 16;
            v8bf plo = *(const v8bf*)pp;
            v8bf phi = *(const v8bf*)(pp + 8);
#pragma unroll
            for (int e = 0; e < 8; ++e) { pf[e] = plo[e]; pf[e + 8] = phi[e]; }
        }
        __syncthreads();
        // ---- O += P @ V ----
#pragma unroll
        for (int t = 0; t < 4; ++t) {
            const __bf16* vrow = Vb + (size_t)(t * 16 + lo16) * JPAD + jb + half * 16;
            v16bf bv = *(const v16bf*)vrow;
            acc[t] = __builtin_amdgcn_wmma_f32_16x16x32_bf16(false, pf, false, bv,
                                                             (short)0, acc[t], false, false);
        }
    }

    // epilogue: 1/l normalize, sigmoid(G+bg) gate, store bf16 [N][512]
#pragma unroll
    for (int t = 0; t < 4; ++t)
#pragma unroll
        for (int r = 0; r < 8; ++r) {
            int q = qbase + r + half * 8;
            int col = h * DHD + t * 16 + lo16;
            float g = Gf[(size_t)q * HD + col] + bg[col];
            float sig = 1.0f / (1.0f + __expf(-g));
            float val = (acc[t][r] / lrun[r]) * sig;
            Ag[(size_t)q * HD + col] = (__bf16)val;
        }
}

// ---------------------------------------------------------------------------
extern "C" void kernel_launch(void* const* d_in, const int* in_sizes, int n_in,
                              void* d_out, int out_size, void* d_ws, size_t ws_size,
                              hipStream_t stream) {
    const float* x     = (const float*)d_in[0];
    const float* Wq    = (const float*)d_in[1];
    const float* Wk    = (const float*)d_in[2];
    const float* Wv    = (const float*)d_in[3];
    const float* Wg    = (const float*)d_in[4];
    const float* bg    = (const float*)d_in[5];
    const float* Wo    = (const float*)d_in[6];
    const float* mem_k = (const float*)d_in[7];
    const float* mem_v = (const float*)d_in[8];
    const float* freqs = (const float*)d_in[9];
    float* out = (float*)d_out;

    char* base = (char*)d_ws;
    size_t off = 0;
    auto carve = [&](size_t bytes) -> char* {
        char* p = base + off;
        off = (off + bytes + 255) & ~(size_t)255;
        return p;
    };
    __bf16* xh  = (__bf16*)carve((size_t)NQ * DMODEL * 2);
    __bf16* WqT = (__bf16*)carve((size_t)DMODEL * HD * 2);
    __bf16* WkT = (__bf16*)carve((size_t)DMODEL * HD * 2);
    __bf16* WvT = (__bf16*)carve((size_t)DMODEL * HD * 2);
    __bf16* WgT = (__bf16*)carve((size_t)DMODEL * HD * 2);
    __bf16* WoT = (__bf16*)carve((size_t)HD * DMODEL * 2);
    float*  Qf  = (float*)carve((size_t)NQ * HD * 4);
    float*  Kr  = (float*)carve((size_t)NQ * HD * 4);
    float*  Vf  = (float*)carve((size_t)NQ * HD * 4);
    float*  Gf  = (float*)carve((size_t)NQ * HD * 4);
    __bf16* QhB = (__bf16*)carve((size_t)HHEADS * NQ * DHD * 2);
    __bf16* KfB = (__bf16*)carve((size_t)HHEADS * JPAD * DHD * 2);
    __bf16* VtB = (__bf16*)carve((size_t)HHEADS * DHD * JPAD * 2);
    __bf16* Ag  = (__bf16*)carve((size_t)NQ * HD * 2);

    // stage 0: convert x, transpose+convert weights
    {
        int n = NQ * DMODEL;
        k_cvt<<<(n + 255) / 256, 256, 0, stream>>>(x, xh, n);
        int w = DMODEL * HD;
        k_cvtT<<<(w + 255) / 256, 256, 0, stream>>>(Wq, WqT, DMODEL, HD);
        k_cvtT<<<(w + 255) / 256, 256, 0, stream>>>(Wk, WkT, DMODEL, HD);
        k_cvtT<<<(w + 255) / 256, 256, 0, stream>>>(Wv, WvT, DMODEL, HD);
        k_cvtT<<<(w + 255) / 256, 256, 0, stream>>>(Wg, WgT, DMODEL, HD);
        k_cvtT<<<(w + 255) / 256, 256, 0, stream>>>(Wo, WoT, HD, DMODEL);
    }
    // stage 1: four projection GEMMs (WMMA bf16, f32 accum, async-LDS B tiles)
    {
        dim3 grid(NQ / 128, HD / 64);
        k_gemm_bf16<<<grid, 256, 0, stream>>>(xh, WqT, Qf, NQ, HD, DMODEL);
        k_gemm_bf16<<<grid, 256, 0, stream>>>(xh, WkT, Kr, NQ, HD, DMODEL);
        k_gemm_bf16<<<grid, 256, 0, stream>>>(xh, WvT, Vf, NQ, HD, DMODEL);
        k_gemm_bf16<<<grid, 256, 0, stream>>>(xh, WgT, Gf, NQ, HD, DMODEL);
    }
    // stage 2: l2norm + rotary + repack; memory tokens
    {
        int n = NQ * HHEADS;
        k_qk_post<<<(n + 255) / 256, 256, 0, stream>>>(Qf, Kr, Vf, freqs, QhB, KfB, VtB);
        k_mem<<<1, 32, 0, stream>>>(mem_k, mem_v, KfB, VtB);
    }
    // stage 3: windowed flash attention + gating (WMMA QK^T and PV)
    {
        dim3 grid(NQ / 64, HHEADS);
        k_attn<<<grid, 128, 0, stream>>>(QhB, KfB, VtB, Gf, bg, Ag);
    }
    // stage 4: output projection GEMM -> d_out (f32)
    {
        dim3 grid(NQ / 128, DMODEL / 64);
        k_gemm_bf16<<<grid, 256, 0, stream>>>(Ag, WoT, out, NQ, DMODEL, HD);
    }
    (void)in_sizes; (void)n_in; (void)out_size; (void)ws_size;
}
